// GatedLinearAttention_20418274525215
// MI455X (gfx1250) — compile-verified, matrix operand order checked by reading, not run on hardware
//
#include <hip/hip_runtime.h>
#include <hip/hip_bf16.h>

// ---------------------------------------------------------------------------
// Gated Linear Attention, CDNA5 (gfx1250, wave32) bf16-WMMA implementation.
//   B=2, S=2048, Hd=2048, H=4, Dk=256, Dv=512, CHUNK=64.
// All WMMA operands are stored K-contiguous in LDS so every fragment load is
// exactly two ds_load_b128 per lane (no scalar ds_load_u16 gather).
// Pipeline:
//   (1) cvt x -> bf16; pack [Wq|Wk|Wv|Wg]^T, Wo^T -> bf16 (K-major weights)
//   (2) gk = logsigmoid(x@Wgk1@Wgk2 + b)/16 ; in-chunk cumsum (in place)
//   (3) QKVG = x @ Wcat    (bf16 WMMA GEMM, async-LDS double-buffered)
//   (4) GLA chunk recurrence (WMMA; S state [e-major 128x260] f32 in LDS)
//   (5) RMS-norm * SiLU(g) gate -> bf16
//   (6) out = gated @ Wo   (bf16 WMMA GEMM)
// ---------------------------------------------------------------------------

typedef __attribute__((ext_vector_type(16))) __bf16 v16bf;
typedef __attribute__((ext_vector_type(8)))  __bf16 v8bf;
typedef __attribute__((ext_vector_type(4)))  __bf16 v4bf;
typedef __attribute__((ext_vector_type(8)))  float  v8f;
typedef __attribute__((ext_vector_type(4)))  int    v4i;

#define BB 2
#define SS 2048
#define HD 2048
#define NH 4
#define DK 256
#define DV 512
#define DV_BLK 128
#define CH 64
#define CHP 72      // padded t-extent (16 lanes -> 16 disjoint bank groups)
#define DKP 260     // padded d-extent for f32 state rows
#define KEYDIM (NH*DK)               // 1024
#define VALDIM (NH*DV)               // 2048
#define NCAT  (KEYDIM*2 + VALDIM*2)  // 6144
#define ROWS  (BB*SS)                // 4096

// ------------------- async global->LDS copy (gfx1250 path) -----------------

#if defined(__HIP_DEVICE_COMPILE__) && \
    __has_builtin(__builtin_amdgcn_global_load_async_to_lds_b128)
#define HAVE_ASYNC 1
#else
#define HAVE_ASYNC 0
#endif

#if HAVE_ASYNC
typedef __attribute__((address_space(1))) v4i* as1_v4i_ptr;  // global int4*
typedef __attribute__((address_space(3))) v4i* as3_v4i_ptr;  // LDS int4*
// 16B global -> LDS, no VGPR roundtrip; tracked with ASYNCcnt.
// ISA 10.2: flat pointers to LDS keep the byte offset in addr[31:0].
__device__ __forceinline__ void async_copy16(__bf16* lds, const __bf16* g) {
  __builtin_amdgcn_global_load_async_to_lds_b128(
      (as1_v4i_ptr)(unsigned long long)g,
      (as3_v4i_ptr)(unsigned)(unsigned long long)lds, 0, 0);
}
#endif

__device__ __forceinline__ void async_wait_all() {
#if HAVE_ASYNC
#if __has_builtin(__builtin_amdgcn_s_wait_asynccnt)
  __builtin_amdgcn_s_wait_asynccnt(0);
#else
  asm volatile("s_wait_asynccnt 0" ::: "memory");
#endif
#endif
}

// ---------------- WMMA fragment helpers (ISA 7.12.2 layouts) ---------------
// For 16-bit A/B fragments, lane (idx = lane&15, hi = lane>=16) holds two
// contiguous K-runs: k in [hi*8, hi*8+8) and [16+hi*8, 16+hi*8+8).
// With a K-contiguous source (element (idx,k) = s[idx*ld + k]) that is two
// 16-byte LDS loads. ld must be a multiple of 8 (byte stride mult of 16).

__device__ __forceinline__ int lane_id() { return threadIdx.x & 31; }

__device__ __forceinline__ v16bf load_frag16(const __bf16* s, int ld) {
  int l = lane_id(), idx = l & 15, hi = l >> 4;
  const __bf16* p = s + idx * ld + hi * 8;
  v8bf lo = *(const v8bf*)p;         // k = hi*8 .. hi*8+7
  v8bf hh = *(const v8bf*)(p + 16);  // k = 16+hi*8 .. 16+hi*8+7
  v16bf f;
#pragma unroll
  for (int j = 0; j < 8; ++j) { f[j] = lo[j]; f[8 + j] = hh[j]; }
  return f;
}

// Same pattern from an f32 source (converted on the fly). ld mult of 4.
__device__ __forceinline__ v16bf load_frag16_f32(const float* s, int ld) {
  int l = lane_id(), idx = l & 15, hi = l >> 4;
  const float* p = s + idx * ld + hi * 8;
  float4 a0 = *(const float4*)p;
  float4 a1 = *(const float4*)(p + 4);
  float4 b0 = *(const float4*)(p + 16);
  float4 b1 = *(const float4*)(p + 20);
  v16bf f;
  f[0]  = (__bf16)a0.x; f[1]  = (__bf16)a0.y; f[2]  = (__bf16)a0.z; f[3]  = (__bf16)a0.w;
  f[4]  = (__bf16)a1.x; f[5]  = (__bf16)a1.y; f[6]  = (__bf16)a1.z; f[7]  = (__bf16)a1.w;
  f[8]  = (__bf16)b0.x; f[9]  = (__bf16)b0.y; f[10] = (__bf16)b0.z; f[11] = (__bf16)b0.w;
  f[12] = (__bf16)b1.x; f[13] = (__bf16)b1.y; f[14] = (__bf16)b1.z; f[15] = (__bf16)b1.w;
  return f;
}

__device__ __forceinline__ v8f wmma_bf16(v16bf a, v16bf b, v8f c) {
  return __builtin_amdgcn_wmma_f32_16x16x32_bf16(
      /*neg_a=*/false, a, /*neg_b=*/false, b,
      /*c_mod=*/(short)0, c, /*reuse_a=*/false, /*reuse_b=*/false);
}

// C/D 16x16 f32: element (m,n): m = vgpr + 8*(lane>=16), n = lane&15
__device__ __forceinline__ void store_c(float* c, int ldc, v8f acc) {
  int l = lane_id(), n = l & 15, m0 = (l >> 4) * 8;
#pragma unroll
  for (int r = 0; r < 8; ++r) c[(m0 + r) * ldc + n] = acc[r];
}

// ---------------------------- small elementwise -----------------------------

__global__ void cvt_bf16_kernel(const float* __restrict__ src,
                                __bf16* __restrict__ dst, int n) {
  for (int i = blockIdx.x * blockDim.x + threadIdx.x; i < n;
       i += gridDim.x * blockDim.x)
    dst[i] = (__bf16)src[i];
}

// transpose-pack: dst[(colofs + c) * ldk + r] = bf16(src[r * cols + c])
// (weights become K-major so GEMM B-tiles are K-contiguous in LDS)
__global__ void pack_wt_kernel(const float* __restrict__ src,
                               __bf16* __restrict__ dst, int rows, int cols,
                               int ldk, int colofs) {
  int n = rows * cols;
  for (int i = blockIdx.x * blockDim.x + threadIdx.x; i < n;
       i += gridDim.x * blockDim.x) {
    int c = i / rows, r = i - c * rows;  // consecutive i -> consecutive r (write-coalesced)
    dst[(size_t)(colofs + c) * ldk + r] = (__bf16)src[(size_t)r * cols + c];
  }
}

// t1 = x @ Wgk1   [ROWS x 16]
__global__ void gk1_kernel(const float* __restrict__ x,
                           const float* __restrict__ w,
                           float* __restrict__ t1) {
  int idx = blockIdx.x * blockDim.x + threadIdx.x;  // ROWS*16
  int row = idx >> 4, col = idx & 15;
  float s = 0.f;
  for (int k = 0; k < HD; ++k) s += x[row * HD + k] * w[k * 16 + col];
  t1[idx] = s;
}

// gk = logsigmoid(t1 @ Wgk2 + b) / 16   [ROWS x 1024]
__global__ void gk2_kernel(const float* __restrict__ t1,
                           const float* __restrict__ w,
                           const float* __restrict__ bias,
                           float* __restrict__ gk) {
  int idx = blockIdx.x * blockDim.x + threadIdx.x;  // ROWS*1024
  int row = idx >> 10, col = idx & 1023;
  float s = bias[col];
#pragma unroll
  for (int k = 0; k < 16; ++k) s += t1[row * 16 + k] * w[k * KEYDIM + col];
  float ls = (s >= 0.f) ? (-log1pf(__expf(-s))) : (s - log1pf(__expf(s)));
  gk[idx] = ls * (1.0f / 16.0f);
}

// In-place cumsum of gk along time within each 64-chunk, per (b,chunk,col).
__global__ void cumsum_kernel(float* __restrict__ gk) {
  int idx = blockIdx.x * blockDim.x + threadIdx.x;  // BB*(SS/CH)*1024
  int col = idx & 1023;
  int bc = idx >> 10;
  int c = bc & (SS / CH - 1), b = bc / (SS / CH);
  int row0 = b * SS + c * CH;
  float acc = 0.f;
  for (int t = 0; t < CH; ++t) {
    float* p = &gk[(row0 + t) * KEYDIM + col];
    acc += *p;
    *p = acc;
  }
}

// --------------------------- bf16 WMMA GEMM --------------------------------
// C[M x N] f32 = A[M x K] bf16 @ Bt[N x K] bf16 (B given transposed/K-major).
// Block tile 128x128, 8 waves, each wave 32x64 (2x4 of 16x16).
// Both LDS tiles are [128 x 32] K-contiguous. Double-buffered async staging.

__device__ __forceinline__ void gemm_stage(const __bf16* __restrict__ A,
                                           const __bf16* __restrict__ Bt,
                                           int K, int bm, int bn, int k0,
                                           int tid, __bf16* dA, __bf16* dB) {
#pragma unroll
  for (int hh = 0; hh < 2; ++hh) {
    int v = tid + hh * 256;  // vector id 0..511 (each vector = 8 bf16 = 16B)
    const __bf16* gA = &A[(size_t)(bm + (v >> 2)) * K + k0 + (v & 3) * 8];
    const __bf16* gB = &Bt[(size_t)(bn + (v >> 2)) * K + k0 + (v & 3) * 8];
    __bf16* lA = &dA[v * 8];
    __bf16* lB = &dB[v * 8];
#if HAVE_ASYNC
    async_copy16(lA, gA);
    async_copy16(lB, gB);
#else
    *(uint4*)lA = *(const uint4*)gA;
    *(uint4*)lB = *(const uint4*)gB;
#endif
  }
}

__global__ void __launch_bounds__(256)
gemm_bf16_kernel(const __bf16* __restrict__ A, const __bf16* __restrict__ Bt,
                 float* __restrict__ C, int M, int N, int K) {
  __shared__ __bf16 sA[2][128 * 32];   // [m x k] K-contiguous
  __shared__ __bf16 sB[2][128 * 32];   // [n x k] K-contiguous
  int bm = blockIdx.y * 128, bn = blockIdx.x * 128;
  int tid = threadIdx.x;
  int w = tid >> 5;
  int wm = (w & 3) * 32;   // wave row offset
  int wn = (w >> 2) * 64;  // wave col offset
  v8f acc[2][4] = {};
  int nk = K >> 5;

  gemm_stage(A, Bt, K, bm, bn, 0, tid, sA[0], sB[0]);
  int cur = 0;
  for (int kt = 0; kt < nk; ++kt) {
    async_wait_all();   // current buffer's loads landed (ASYNCcnt == 0)
    __syncthreads();    // ... in every wave
    if (kt + 1 < nk)    // overlap: next tile streams in while we do WMMA
      gemm_stage(A, Bt, K, bm, bn, (kt + 1) << 5, tid, sA[cur ^ 1],
                 sB[cur ^ 1]);
    v16bf af[2], bfr[4];
#pragma unroll
    for (int i = 0; i < 2; ++i)
      af[i] = load_frag16(&sA[cur][(wm + i * 16) * 32], 32);
#pragma unroll
    for (int j = 0; j < 4; ++j)
      bfr[j] = load_frag16(&sB[cur][(wn + j * 16) * 32], 32);
#pragma unroll
    for (int i = 0; i < 2; ++i)
#pragma unroll
      for (int j = 0; j < 4; ++j)
        acc[i][j] = wmma_bf16(af[i], bfr[j], acc[i][j]);
    cur ^= 1;
  }
#pragma unroll
  for (int i = 0; i < 2; ++i)
#pragma unroll
    for (int j = 0; j < 4; ++j)
      store_c(&C[(size_t)(bm + wm + i * 16) * N + bn + wn + j * 16], N,
              acc[i][j]);
}

// ------------------------------- GLA core ----------------------------------
// One block per (b, h, e_blk). State S (e-major [128 x DKP] f32) in LDS across
// the 32 sequential chunks. Per chunk:
//   Qg = q*exp(g)/sqrt(Dk), Kg = k*exp(-g)  (bf16; Kg also kept transposed)
//   A  = tril(Qg Kg^T)                      (WMMA, bf16 in LDS)
//   O  = A V + Qg S                         (WMMA, f32 to global)
//   S  = exp(g_last) (x) (S + Kg^T V)       (WMMA accumulate + row scale)

constexpr int GLA_LDS = DV_BLK * DKP * 4     // sS   133120 (e-major, d-contig)
                      + CH * DK * 2          // sQg   32768 (row-major)
                      + CH * DK * 2          // sKg   32768 (row-major)
                      + DK * CHP * 2         // sKgT  36864 (d-major, t-contig)
                      + DV_BLK * CHP * 2     // sVt   18432 (e-major, t-contig)
                      + CH * CH * 2          // sA     8192 (row-major)
                      + DK * 4;              // sEg    1024   => 263168 B

__global__ void __launch_bounds__(256)
gla_kernel(const float* __restrict__ qkvg,  // [ROWS x 6144]
           const float* __restrict__ gc,    // [ROWS x 1024] cumsum'd in chunk
           float* __restrict__ O) {         // [ROWS x 2048]
  extern __shared__ char smem[];
  float*  sS   = (float*)smem;                              // [DV_BLK][DKP]
  __bf16* sQg  = (__bf16*)(smem + DV_BLK * DKP * 4);        // [CH][DK]
  __bf16* sKg  = sQg + CH * DK;                             // [CH][DK]
  __bf16* sKgT = sKg + CH * DK;                             // [DK][CHP]
  __bf16* sVt  = sKgT + DK * CHP;                           // [DV_BLK][CHP]
  __bf16* sA   = sVt + DV_BLK * CHP;                        // [CH][CH]
  float*  sEg  = (float*)(sA + CH * CH);                    // [DK]

  int blk = blockIdx.x;           // 0..31
  int e_blk = blk & 3;
  int h = (blk >> 2) & 3;
  int b = blk >> 4;
  int tid = threadIdx.x, w = tid >> 5, l = tid & 31;
  const float scale = 0.0625f;    // 256^-0.5

  for (int i = tid; i < DV_BLK * DKP; i += 256) sS[i] = 0.f;

  for (int c = 0; c < SS / CH; ++c) {
    __syncthreads();  // prior chunk done with staging buffers
    int s0 = c * CH;

    // ---- stage Qg / Kg (+ transposed Kg) ----
    for (int i = tid; i < (CH * DK) / 4; i += 256) {
      int e0 = i * 4;
      int t = e0 >> 8, d = e0 & 255;
      int row = b * SS + s0 + t;
      float4 g4 = *(const float4*)&gc[row * KEYDIM + h * DK + d];
      float4 q4 = *(const float4*)&qkvg[(size_t)row * NCAT + h * DK + d];
      float4 k4 = *(const float4*)&qkvg[(size_t)row * NCAT + KEYDIM + h * DK + d];
      const float* gs = (const float*)&g4;
      const float* qs = (const float*)&q4;
      const float* ks = (const float*)&k4;
      v4bf qb, kb;
#pragma unroll
      for (int j = 0; j < 4; ++j) {
        qb[j] = (__bf16)(qs[j] * __expf(gs[j]) * scale);
        kb[j] = (__bf16)(ks[j] * __expf(-gs[j]));
      }
      *(v4bf*)&sQg[t * DK + d] = qb;
      *(v4bf*)&sKg[t * DK + d] = kb;
#pragma unroll
      for (int j = 0; j < 4; ++j) sKgT[(d + j) * CHP + t] = kb[j];
    }
    // ---- stage V transposed: sVt[e][t] (t-contiguous) ----
    for (int i = tid; i < DV_BLK * (CH / 4); i += 256) {
      int e = i & (DV_BLK - 1);
      int t = (i >> 7) * 4;
      const float* src = &qkvg[(size_t)(b * SS + s0 + t) * NCAT + 2 * KEYDIM +
                               h * DV + e_blk * DV_BLK + e];
      v4bf vb;
#pragma unroll
      for (int j = 0; j < 4; ++j) vb[j] = (__bf16)src[(size_t)j * NCAT];
      *(v4bf*)&sVt[e * CHP + t] = vb;
    }
    // ---- exp(g_last) ----
    {
      int row = b * SS + s0 + (CH - 1);
      sEg[tid] = __expf(gc[row * KEYDIM + h * DK + tid]);
    }
    __syncthreads();

    // ---- A = tril(Qg Kg^T), 16 tiles of 16x16, 2 per wave ----
#pragma unroll
    for (int t2 = 0; t2 < 2; ++t2) {
      int tile = w * 2 + t2;
      int ti = tile >> 2, tj = tile & 3;
      v8f a = {};
      if (tj <= ti) {  // wave-uniform branch: EXEC stays all-ones for WMMA
#pragma unroll
        for (int kk = 0; kk < DK; kk += 32) {
          v16bf af = load_frag16(&sQg[ti * 16 * DK + kk], DK);
          v16bf bf = load_frag16(&sKg[tj * 16 * DK + kk], DK);  // B=Kg^T
          a = wmma_bf16(af, bf, a);
        }
      }
      int n = l & 15, m0 = (l >> 4) * 8;
#pragma unroll
      for (int r = 0; r < 8; ++r) {
        int gm = ti * 16 + m0 + r, gn = tj * 16 + n;
        sA[gm * CH + gn] = (__bf16)((gn <= gm) ? a[r] : 0.f);
      }
    }
    __syncthreads();

    // ---- O = A V + Qg S : 32 tiles (4 x 8), 4 per wave ----
    v8f oacc[4];
    int oti[4], otj[4];
#pragma unroll
    for (int t4 = 0; t4 < 4; ++t4) {
      int tile = w + t4 * 8;
      int ti = tile >> 3, tj = tile & 7;
      oti[t4] = ti; otj[t4] = tj;
      v8f acc = {};
#pragma unroll
      for (int kk = 0; kk < CH; kk += 32) {   // intra-chunk: A @ V
        v16bf af = load_frag16(&sA[ti * 16 * CH + kk], CH);
        v16bf bf = load_frag16(&sVt[(tj * 16) * CHP + kk], CHP);
        acc = wmma_bf16(af, bf, acc);
      }
#pragma unroll
      for (int kk = 0; kk < DK; kk += 32) {   // inter-chunk: Qg @ S
        v16bf af = load_frag16(&sQg[ti * 16 * DK + kk], DK);
        v16bf bf = load_frag16_f32(&sS[(tj * 16) * DKP + kk], DKP);
        acc = wmma_bf16(af, bf, acc);
      }
      oacc[t4] = acc;
    }
#pragma unroll
    for (int t4 = 0; t4 < 4; ++t4) {
      int n = l & 15, m0 = (l >> 4) * 8;
#pragma unroll
      for (int r = 0; r < 8; ++r) {
        int row = b * SS + s0 + oti[t4] * 16 + m0 + r;
        int col = h * DV + e_blk * DV_BLK + otj[t4] * 16 + n;
        O[(size_t)row * VALDIM + col] = oacc[t4][r];
      }
    }
    __syncthreads();  // all reads of sS complete before update

    // ---- S = exp(g_last) (x) (S + Kg^T V) : 128 tiles, 16 per wave ----
#pragma unroll
    for (int t16 = 0; t16 < 16; ++t16) {
      int tile = w * 16 + t16;
      int di = tile >> 3, ej = tile & 7;
      int n = l & 15, m0 = (l >> 4) * 8;
      // C fragment: e = ej*16+n (lane), d = di*16+m0+r (contiguous in memory)
      float* cptr = &sS[(size_t)(ej * 16 + n) * DKP + di * 16 + m0];
      float4 c0 = *(float4*)cptr;
      float4 c1 = *(float4*)(cptr + 4);
      v8f sc;
      sc[0] = c0.x; sc[1] = c0.y; sc[2] = c0.z; sc[3] = c0.w;
      sc[4] = c1.x; sc[5] = c1.y; sc[6] = c1.z; sc[7] = c1.w;
#pragma unroll
      for (int kk = 0; kk < CH; kk += 32) {
        v16bf af = load_frag16(&sKgT[(di * 16) * CHP + kk], CHP);  // Kg^T
        v16bf bf = load_frag16(&sVt[(ej * 16) * CHP + kk], CHP);   // V
        sc = wmma_bf16(af, bf, sc);
      }
#pragma unroll
      for (int r = 0; r < 8; ++r) sc[r] *= sEg[di * 16 + m0 + r];
      c0.x = sc[0]; c0.y = sc[1]; c0.z = sc[2]; c0.w = sc[3];
      c1.x = sc[4]; c1.y = sc[5]; c1.z = sc[6]; c1.w = sc[7];
      *(float4*)cptr = c0;
      *(float4*)(cptr + 4) = c1;
    }
  }
}

// ------------------------- RMS norm + SiLU gate ----------------------------
// One block per (row, head). o' = o / sqrt(mean(o^2)+eps) * w * g*sigmoid(g)

__global__ void __launch_bounds__(256)
gate_kernel(const float* __restrict__ O, const float* __restrict__ qkvg,
            const float* __restrict__ gnw, __bf16* __restrict__ og) {
  __shared__ float red[9];
  int rh = blockIdx.x;
  int h = rh & 3, row = rh >> 2;
  int tid = threadIdx.x;
  float o0 = O[(size_t)row * VALDIM + h * DV + tid * 2];
  float o1 = O[(size_t)row * VALDIM + h * DV + tid * 2 + 1];
  float ss = o0 * o0 + o1 * o1;
#pragma unroll
  for (int off = 16; off; off >>= 1) ss += __shfl_down(ss, off, 32);
  if ((tid & 31) == 0) red[tid >> 5] = ss;
  __syncthreads();
  if (tid == 0) {
    float t = 0.f;
#pragma unroll
    for (int i = 0; i < 8; ++i) t += red[i];
    red[8] = rsqrtf(t * (1.0f / (float)DV) + 1e-5f);
  }
  __syncthreads();
  float rinv = red[8];
#pragma unroll
  for (int j = 0; j < 2; ++j) {
    int e = tid * 2 + j;
    float o = j ? o1 : o0;
    float g = qkvg[(size_t)row * NCAT + 2 * KEYDIM + VALDIM + h * DV + e];
    float silu = g / (1.f + __expf(-g));
    og[(size_t)row * VALDIM + h * DV + e] = (__bf16)(o * rinv * gnw[e] * silu);
  }
}

// ------------------------------ launcher -----------------------------------

extern "C" void kernel_launch(void* const* d_in, const int* in_sizes, int n_in,
                              void* d_out, int out_size, void* d_ws,
                              size_t ws_size, hipStream_t stream) {
  const float* x    = (const float*)d_in[0];  // [4096 x 2048]
  const float* Wq   = (const float*)d_in[1];  // [2048 x 1024]
  const float* Wk   = (const float*)d_in[2];
  const float* Wv   = (const float*)d_in[3];  // [2048 x 2048]
  const float* Wgk1 = (const float*)d_in[4];  // [2048 x 16]
  const float* Wgk2 = (const float*)d_in[5];  // [16 x 1024]
  const float* bgk2 = (const float*)d_in[6];  // [1024]
  const float* Wg   = (const float*)d_in[7];  // [2048 x 2048]
  const float* gnw  = (const float*)d_in[8];  // [512]
  const float* Wo   = (const float*)d_in[9];  // [2048 x 2048]
  float* out = (float*)d_out;                 // [4096 x 2048]

  // workspace carve-up (bytes)
  char* ws = (char*)d_ws;
  __bf16* xb   = (__bf16*)(ws);                              // 16,777,216
  __bf16* wcat = (__bf16*)(ws + 16777216);  // [6144 x 2048] K-major, 25,165,824
  __bf16* wob  = (__bf16*)(ws + 41943040);  // [2048 x 2048] K-major,  8,388,608
  float*  qkvg = (float*)(ws + 50331648);                    // 100,663,296
  float*  t1   = (float*)(ws + 150994944);                   //    262,144
  float*  gk   = (float*)(ws + 151257088);                   // 16,777,216
  float*  Obuf = (float*)(ws + 168034304);                   // 33,554,432
  __bf16* og   = (__bf16*)(ws + 201588736);                  // 16,777,216
  (void)in_sizes; (void)n_in; (void)out_size; (void)ws_size; // ~208 MB total

  // (1) conversions / transposed weight packing (K-major)
  cvt_bf16_kernel<<<8192, 256, 0, stream>>>(x, xb, ROWS * HD);
  pack_wt_kernel<<<4096, 256, 0, stream>>>(Wq, wcat, HD, KEYDIM, HD, 0);
  pack_wt_kernel<<<4096, 256, 0, stream>>>(Wk, wcat, HD, KEYDIM, HD, KEYDIM);
  pack_wt_kernel<<<8192, 256, 0, stream>>>(Wv, wcat, HD, VALDIM, HD, 2 * KEYDIM);
  pack_wt_kernel<<<8192, 256, 0, stream>>>(Wg, wcat, HD, VALDIM, HD, 2 * KEYDIM + VALDIM);
  pack_wt_kernel<<<8192, 256, 0, stream>>>(Wo, wob, HD, HD, HD, 0);

  // (2) gk path + in-chunk cumsum (in place on gk)
  gk1_kernel<<<(ROWS * 16) / 256, 256, 0, stream>>>(x, Wgk1, t1);
  gk2_kernel<<<(ROWS * KEYDIM) / 256, 256, 0, stream>>>(t1, Wgk2, bgk2, gk);
  cumsum_kernel<<<(BB * (SS / CH) * KEYDIM) / 256, 256, 0, stream>>>(gk);

  // (3) QKVG = x @ [Wq|Wk|Wv|Wg]   (4096 x 6144)
  gemm_bf16_kernel<<<dim3(NCAT / 128, ROWS / 128), 256, 0, stream>>>(
      xb, wcat, qkvg, ROWS, NCAT, HD);

  // (4) GLA recurrence: 32 blocks = (b, h, Dv/128)
  gla_kernel<<<BB * NH * (DV / DV_BLK), 256, GLA_LDS, stream>>>(qkvg, gk, Obuf);

  // (5) RMS-norm + SiLU gating -> bf16
  gate_kernel<<<ROWS * NH, 256, 0, stream>>>(Obuf, qkvg, gnw, og);

  // (6) out = og @ Wo   (4096 x 2048)
  gemm_bf16_kernel<<<dim3(HD / 128, ROWS / 128), 256, 0, stream>>>(
      og, wob, out, ROWS, HD, HD);
}
